// DeformConv3d_88201448391297
// MI455X (gfx1250) — matrix operand order repaired and checked by reading, not compiled, not run on hardware
//
#include <hip/hip_runtime.h>

// ---------------- static problem config (from reference) ----------------
#define NB    2
#define CINC  64
#define COUTC 128
#define DD    16
#define HH    28
#define WW    28
#define SPA   (DD*HH*WW)      // 12544 spatial per sample
#define K3    27
#define KDIM  (CINC*K3)       // 1728 GEMM K
#define NPOS  (NB*SPA)        // 25088 GEMM N
#define OFFC  (3*K3)          // 81 offset channels
#define OFFM  96              // 81 padded up to 6 WMMA M-tiles

typedef __attribute__((ext_vector_type(16))) __bf16 v16bf;
typedef __attribute__((ext_vector_type(8)))  __bf16 v8bf;
typedef __attribute__((ext_vector_type(8)))  float  v8f;

static __device__ __forceinline__ unsigned short f32_to_bf16_rne(float f) {
    union { float f; unsigned u; } v; v.f = f;
    unsigned u = v.u;
    unsigned r = (u + 0x7FFFu + ((u >> 16) & 1u)) >> 16;   // round-to-nearest-even
    return (unsigned short)r;
}

// ---------------- weight conversion kernels ------------------------------
__global__ void __launch_bounds__(256)
wcvt_kernel(const float* __restrict__ w, unsigned short* __restrict__ wbf) {
    int i = blockIdx.x * 256 + threadIdx.x;
    if (i < COUTC * KDIM) wbf[i] = f32_to_bf16_rne(w[i]);
}

// offset-conv weights: 81 real rows, pad to 96 rows of zeros
__global__ void __launch_bounds__(256)
wcvt_off_kernel(const float* __restrict__ w, unsigned short* __restrict__ wbf) {
    int i = blockIdx.x * 256 + threadIdx.x;
    if (i >= OFFM * KDIM) return;
    int row = i / KDIM;
    wbf[i] = (row < OFFC) ? f32_to_bf16_rne(w[i]) : (unsigned short)0;
}

// ---- kernel A: offset conv as im2col -> LDS -> WMMA implicit GEMM -------
// M=96 (6 waves of the 8-wave WG do WMMA), K=1728, N-tile=16 positions.
__global__ void __launch_bounds__(256)
off_gemm_kernel(const float* __restrict__ x, const unsigned short* __restrict__ wbf,
                const float* __restrict__ bias, float* __restrict__ off) {
    __shared__ __align__(32) unsigned short Bs[16 * KDIM];   // 54 KB

    const int tid   = threadIdx.x;
    const int pbase = blockIdx.x * 16;

    // ---- stage: im2col of x into LDS (zero padding outside) ----
    for (int idx = tid; idx < 16 * K3; idx += 256) {
        int pos = idx / K3, k3 = idx % K3;
        int p = pbase + pos;
        int n = p / SPA, sp = p % SPA;
        int dz = sp / (HH*WW), hy = (sp / WW) % HH, wx = sp % WW;
        int kd = k3 / 9, kh = (k3 / 3) % 3, kw = k3 % 3;

        int d = dz + kd - 1, h = hy + kh - 1, ww2 = wx + kw - 1;
        bool ok = ((unsigned)d < (unsigned)DD) && ((unsigned)h < (unsigned)HH)
               && ((unsigned)ww2 < (unsigned)WW);
        int lin = ok ? d*(HH*WW) + h*WW + ww2 : 0;

        const float* xn = x + (size_t)n * CINC * SPA;
        unsigned short* bsrow = Bs + pos * KDIM + k3;
        for (int c = 0; c < CINC; ++c) {
            float v = ok ? xn[c * SPA + lin] : 0.f;
            bsrow[c * K3] = f32_to_bf16_rne(v);
        }
    }
    __syncthreads();

    // ---- WMMA phase: waves 0..5 each do 16(M) x 16(N), K=1728/32 steps
    const int wave = tid >> 5;
    if (wave >= OFFM / 16) return;
    const int lane  = tid & 31;
    const int half  = lane >> 4;
    const int l16   = lane & 15;
    const int mbase = wave * 16;

    const unsigned short* Arow = wbf + (size_t)(mbase + l16) * KDIM;
    const unsigned short* Brow = Bs  + l16 * KDIM;

    v8f acc = {0.f, 0.f, 0.f, 0.f, 0.f, 0.f, 0.f, 0.f};
    for (int kk = 0; kk < KDIM; kk += 32) {
        v8bf alo = *(const v8bf*)(Arow + kk + 8 * half);
        v8bf ahi = *(const v8bf*)(Arow + kk + 16 + 8 * half);
        v16bf a = __builtin_shufflevector(alo, ahi,
                    0,1,2,3,4,5,6,7,8,9,10,11,12,13,14,15);
        v8bf blo = *(const v8bf*)(Brow + kk + 8 * half);
        v8bf bhi = *(const v8bf*)(Brow + kk + 16 + 8 * half);
        v16bf b = __builtin_shufflevector(blo, bhi,
                    0,1,2,3,4,5,6,7,8,9,10,11,12,13,14,15);
        acc = __builtin_amdgcn_wmma_f32_16x16x32_bf16(
                  false, a, false, b, (short)0, acc, false, false);
    }

    // ---- write-back: off[n][m][sp] = acc + bias[m]  (skip padded rows)
    int p = pbase + l16;
    int n = p / SPA, sp = p % SPA;
    float* ob = off + (size_t)n * OFFC * SPA + sp;
    #pragma unroll
    for (int r = 0; r < 8; ++r) {
        int m = mbase + r + 8 * half;
        if (m < OFFC) ob[(size_t)m * SPA] = acc[r] + bias[m];
    }
}

// ---- kernel B: fused trilinear gather -> LDS -> WMMA implicit GEMM ------
// Workgroup: 256 threads (8 wave32). Tile: M=128 x N=16 positions.
__global__ void __launch_bounds__(256)
deform_gemm_kernel(const float* __restrict__ x, const float* __restrict__ off,
                   const unsigned short* __restrict__ wbf, float* __restrict__ y) {
    __shared__ __align__(32) unsigned short Bs[16 * KDIM];   // 54 KB

    const int tid   = threadIdx.x;
    const int pbase = blockIdx.x * 16;

    // ---- stage: trilinear sampling of V[c*27+k3, pos] into LDS ----
    for (int idx = tid; idx < 16 * K3; idx += 256) {
        int pos = idx / K3, k3 = idx % K3;
        int p = pbase + pos;
        int n = p / SPA, sp = p % SPA;
        int dz = sp / (HH*WW), hy = (sp / WW) % HH, wx = sp % WW;
        int kd = k3 / 9, kh = (k3 / 3) % 3, kw = k3 % 3;

        const float* ob = off + (size_t)n * OFFC * SPA + sp;
        float pd = ob[(0*K3 + k3) * SPA] + (float)(kd + dz);   // padded coords
        float ph = ob[(1*K3 + k3) * SPA] + (float)(kh + hy);
        float pw = ob[(2*K3 + k3) * SPA] + (float)(kw + wx);

        float fd = pd - floorf(pd), fh = ph - floorf(ph), fw = pw - floorf(pw);
        int d0 = (int)floorf(pd) - 1;     // unpadded base (pad=1)
        int h0 = (int)floorf(ph) - 1;
        int w0 = (int)floorf(pw) - 1;

        float corw[8]; int corl[8];
        #pragma unroll
        for (int a = 0; a < 2; ++a)
        #pragma unroll
        for (int bb = 0; bb < 2; ++bb)
        #pragma unroll
        for (int cc = 0; cc < 2; ++cc) {
            int i  = a*4 + bb*2 + cc;
            int id = d0 + a, ih = h0 + bb, iw = w0 + cc;
            bool ok = ((unsigned)id < (unsigned)DD) && ((unsigned)ih < (unsigned)HH)
                   && ((unsigned)iw < (unsigned)WW);
            float wv = (a ? fd : 1.f - fd) * (bb ? fh : 1.f - fh) * (cc ? fw : 1.f - fw);
            corw[i] = ok ? wv : 0.f;
            corl[i] = ok ? id*(HH*WW) + ih*WW + iw : 0;
        }

        const float* xn = x + (size_t)n * CINC * SPA;
        unsigned short* bsrow = Bs + pos * KDIM + k3;
        for (int c = 0; c < CINC; ++c) {
            const float* xc = xn + c * SPA;
            float v = 0.f;
            #pragma unroll
            for (int i = 0; i < 8; ++i) v += corw[i] * xc[corl[i]];
            bsrow[c * K3] = f32_to_bf16_rne(v);
        }
    }
    __syncthreads();

    // ---- WMMA phase: each wave does 16(M) x 16(N), K=1728 in steps of 32
    const int wave  = tid >> 5;
    const int lane  = tid & 31;
    const int half  = lane >> 4;     // K-half selector (A) / K-group (B)
    const int l16   = lane & 15;     // M row (A) / N col (B,C,D)
    const int mbase = wave * 16;

    const unsigned short* Arow = wbf + (size_t)(mbase + l16) * KDIM;
    const unsigned short* Brow = Bs  + l16 * KDIM;

    v8f acc = {0.f, 0.f, 0.f, 0.f, 0.f, 0.f, 0.f, 0.f};
    for (int kk = 0; kk < KDIM; kk += 32) {
        // A 16x32 bf16: elems 0..7 = K[8h..8h+7], elems 8..15 = K[16+8h..+7]
        v8bf alo = *(const v8bf*)(Arow + kk + 8 * half);
        v8bf ahi = *(const v8bf*)(Arow + kk + 16 + 8 * half);
        v16bf a = __builtin_shufflevector(alo, ahi,
                    0,1,2,3,4,5,6,7,8,9,10,11,12,13,14,15);
        // B 32x16 bf16: elems 0..7 = K[8h..8h+7], elems 8..15 = K[16+8h..+7] at column l16
        v8bf blo = *(const v8bf*)(Brow + kk + 8 * half);
        v8bf bhi = *(const v8bf*)(Brow + kk + 16 + 8 * half);
        v16bf b = __builtin_shufflevector(blo, bhi,
                    0,1,2,3,4,5,6,7,8,9,10,11,12,13,14,15);
        acc = __builtin_amdgcn_wmma_f32_16x16x32_bf16(
                  false, a, false, b, (short)0, acc, false, false);
    }

    // ---- write-back pre-BN y: C/D layout: VGPR r -> M=r+8*half, N=l16
    int p = pbase + l16;
    int n = p / SPA, sp = p % SPA;
    float* yb = y + (size_t)n * COUTC * SPA + sp;
    #pragma unroll
    for (int r = 0; r < 8; ++r) {
        int m = mbase + r + 8 * half;
        yb[(size_t)m * SPA] = acc[r];
    }
}

// ---------------- kernel C1: per-channel BN statistics -------------------
__global__ void __launch_bounds__(256)
bn_stats_kernel(const float* __restrict__ y, float* __restrict__ mean,
                float* __restrict__ inv) {
    __shared__ float ssum[256], ssq[256];
    int m = blockIdx.x, tid = threadIdx.x;
    float s = 0.f, q = 0.f;
    for (int i = tid; i < NPOS; i += 256) {
        int n = i / SPA, sp = i % SPA;
        float v = y[(size_t)(n * COUTC + m) * SPA + sp];
        s += v; q += v * v;
    }
    ssum[tid] = s; ssq[tid] = q;
    __syncthreads();
    for (int st = 128; st > 0; st >>= 1) {
        if (tid < st) { ssum[tid] += ssum[tid + st]; ssq[tid] += ssq[tid + st]; }
        __syncthreads();
    }
    if (tid == 0) {
        float mu  = ssum[0] * (1.f / (float)NPOS);
        float var = ssq[0] * (1.f / (float)NPOS) - mu * mu;
        mean[m] = mu;
        inv[m]  = rsqrtf(var + 1e-5f);
    }
}

// ---------------- kernel C2: normalize + affine + ReLU (in place) --------
__global__ void __launch_bounds__(256)
bn_relu_kernel(float* __restrict__ y, const float* __restrict__ mean,
               const float* __restrict__ inv, const float* __restrict__ gamma,
               const float* __restrict__ beta) {
    int i = blockIdx.x * 256 + threadIdx.x;
    if (i >= NB * COUTC * SPA) return;
    int m = (i / SPA) % COUTC;
    float v = (y[i] - mean[m]) * inv[m] * gamma[m] + beta[m];
    y[i] = v > 0.f ? v : 0.f;
}

// ---------------- launcher ----------------------------------------------
extern "C" void kernel_launch(void* const* d_in, const int* in_sizes, int n_in,
                              void* d_out, int out_size, void* d_ws, size_t ws_size,
                              hipStream_t stream) {
    const float* x      = (const float*)d_in[0];
    const float* w_off  = (const float*)d_in[1];
    const float* b_off  = (const float*)d_in[2];
    const float* w_conv = (const float*)d_in[3];
    const float* gamma  = (const float*)d_in[4];
    const float* beta   = (const float*)d_in[5];
    float* y = (float*)d_out;

    // workspace carve-up (all offsets 32B aligned)
    const size_t off_bytes  = (size_t)NB * OFFC * SPA * sizeof(float);         // 8,128,512
    const size_t wbf_bytes  = (size_t)COUTC * KDIM * sizeof(unsigned short);   //   442,368
    const size_t wofb_bytes = (size_t)OFFM * KDIM * sizeof(unsigned short);    //   331,776
    float*          off_ws  = (float*)d_ws;
    unsigned short* wbf     = (unsigned short*)((char*)d_ws + off_bytes);
    unsigned short* wbf_off = (unsigned short*)((char*)d_ws + off_bytes + wbf_bytes);
    float*          mean    = (float*)((char*)d_ws + off_bytes + wbf_bytes + wofb_bytes);
    float*          inv     = mean + COUTC;

    // 1) weight bf16 conversions (independent)
    wcvt_kernel<<<(COUTC * KDIM + 255) / 256, 256, 0, stream>>>(w_conv, wbf);
    wcvt_off_kernel<<<(OFFM * KDIM + 255) / 256, 256, 0, stream>>>(w_off, wbf_off);

    // 2) offset conv as WMMA implicit GEMM (im2col in LDS)
    off_gemm_kernel<<<NPOS / 16, 256, 0, stream>>>(x, wbf_off, b_off, off_ws);

    // 3) fused deformable implicit GEMM (WMMA), y = pre-BN output
    deform_gemm_kernel<<<NPOS / 16, 256, 0, stream>>>(x, off_ws, wbf, y);

    // 4) BN batch statistics
    bn_stats_kernel<<<COUTC, 256, 0, stream>>>(y, mean, inv);

    // 5) normalize + affine + ReLU in place
    bn_relu_kernel<<<(NB * COUTC * SPA + 255) / 256, 256, 0, stream>>>(y, mean, inv, gamma, beta);
}
